// conv_block_Condconv_57397942944186
// MI455X (gfx1250) — compile-verified, matrix operand order checked by reading, not run on hardware
//
#include <hip/hip_runtime.h>

// ---------------------------------------------------------------------------
// CondConv x2 (conv3x3 pad1) + BN + ReLU for MI455X (gfx1250, wave32, WMMA).
//
// Implicit GEMM with V_WMMA_F32_16X16X32_BF16:
//   M = output pixels (2 rows x 32 cols per block -> 4 16x16 M-tiles/wave),
//   N = Cout (128 = 8 waves x 16), K = 9 taps * Cin ordered (tap, ci) so each
//   32-wide K chunk fixes the filter tap.
// Activations staged as bf16 halo tiles in LDS (coalesced global reads,
// transposed to ci-contiguous on the LDS side). Per-sample mixed weights are
// pre-shuffled into exact WMMA B-fragment order, so the conv inner loop's
// weight load is one contiguous 32B/lane stream from L2.
// fp32 accumulate, fused bias+BN+ReLU epilogue. Intermediate kept bf16 in ws.
// ---------------------------------------------------------------------------

typedef __bf16 bf16;
typedef __attribute__((ext_vector_type(16))) __bf16 v16bf;
typedef __attribute__((ext_vector_type(8)))  __bf16 v8bf;
typedef __attribute__((ext_vector_type(8)))  float  v8f;
typedef __attribute__((ext_vector_type(4)))  float  v4f;

#define BATCH 16
#define COUT  128
#define HH    128
#define WW    128
#define NEXP  3
#define BN_EPS 1e-5f

// ---------------------------------------------------------------------------
// 1) Global average pool over HxW: one block per (b, c).
// ---------------------------------------------------------------------------
template <typename T>
__global__ __launch_bounds__(256) void pool_mean_kernel(
    const T* __restrict__ in, float* __restrict__ pooled) {
  const int HW = HH * WW;
  const T* p = in + (size_t)blockIdx.x * HW;
  float s = 0.0f;
  for (int i = threadIdx.x; i < HW; i += 256) s += (float)p[i];
  __shared__ float red[256];
  red[threadIdx.x] = s;
  __syncthreads();
  for (int off = 128; off > 0; off >>= 1) {
    if ((int)threadIdx.x < off) red[threadIdx.x] += red[threadIdx.x + off];
    __syncthreads();
  }
  if (threadIdx.x == 0) pooled[blockIdx.x] = red[0] * (1.0f / HW);
}

// ---------------------------------------------------------------------------
// 2) Routing: rw[b,e] = sigmoid(pooled[b,:] . fcw[e,:] + fcb[e])   (tiny)
// ---------------------------------------------------------------------------
__global__ void routing_kernel(const float* __restrict__ pooled,
                               const float* __restrict__ fcw,
                               const float* __restrict__ fcb,
                               float* __restrict__ rw, int C) {
  int t = threadIdx.x;
  if (t < BATCH * NEXP) {
    int b = t / NEXP, e = t % NEXP;
    float z = fcb[e];
    for (int c = 0; c < C; ++c) z += pooled[b * C + c] * fcw[e * C + c];
    rw[t] = 1.0f / (1.0f + expf(-z));
  }
}

// ---------------------------------------------------------------------------
// 3) Per-sample expert mix -> bf16 weights in WMMA B-fragment order:
//      wmix[(((b*9+tap)*(CIN/32)+cb)*8 + wave)*32 + lane][16 halves]
//    lane = g*16+n (n = cout%16, g = K-half group), halves: [hi*8+j] with
//      K offset = g*8 + j (hi=0)  /  16 + g*8 + j (hi=1)     (ISA 7.12.2)
//    Source w layout: [E][Cout][Cin][3][3]  (tap innermost).
// ---------------------------------------------------------------------------
__global__ __launch_bounds__(256) void mix_weights_kernel(
    const float* __restrict__ w, const float* __restrict__ rw,
    bf16* __restrict__ wmix, int CIN) {
  size_t total = (size_t)BATCH * COUT * 9 * CIN;
  for (size_t idx = (size_t)blockIdx.x * 256 + threadIdx.x; idx < total;
       idx += (size_t)gridDim.x * 256) {
    int ci  = (int)(idx % CIN);
    int tap = (int)((idx / CIN) % 9);
    int co  = (int)((idx / ((size_t)CIN * 9)) % COUT);
    int b   = (int)(idx / ((size_t)CIN * 9 * COUT));
    float acc = 0.0f;
#pragma unroll
    for (int e = 0; e < NEXP; ++e)
      acc += rw[b * NEXP + e] *
             w[(((size_t)e * COUT + co) * CIN + ci) * 9 + tap];
    // fragment-order destination
    int wave = co >> 4, n = co & 15;
    int cb = ci >> 5, r = ci & 31;
    int g  = (r >> 3) & 1;   // K-half lane group
    int hi = (r >> 4) & 1;   // low/high 8 halves within the v16bf
    int j  = r & 7;
    size_t dst =
        ((((size_t)(b * 9 + tap) * (CIN / 32) + cb) * 8 + wave) * 32 +
         (g * 16 + n)) * 16 + hi * 8 + j;
    wmix[dst] = (bf16)acc;
  }
}

// ---------------------------------------------------------------------------
// 4) Implicit-GEMM conv3x3 + bias + BN + ReLU via WMMA bf16.
//    Block = 256 thr = 8 waves; each wave owns 16 couts (8*16 = 128 = Cout).
//    Block computes 2 rows x 32 pixels: 4 16x16 WMMA M-tiles per wave.
//    LDS: halo tile xs[4][34][CINP] bf16 (rows h0-1 .. h0+2).
//    A-frag (ISA 16-bit A 16x32): lane m=lane&15, group g=lane>>4;
//      halves 0..7  -> ci = c0 + g*8 .. +7,  halves 8..15 -> ci = c0+16+g*8..
//    C/D (f32 16x16): vgpr v -> M = v + g*8, lane -> N.
// ---------------------------------------------------------------------------
template <int CIN, bool IN_F32, bool OUT_BF16>
__global__ __launch_bounds__(256) void condconv_wmma_kernel(
    const void* __restrict__ in_v, const bf16* __restrict__ wmix,
    const float* __restrict__ bias,
    const float* __restrict__ bng, const float* __restrict__ bnb,
    const float* __restrict__ bnm, const float* __restrict__ bnv,
    void* __restrict__ out_v) {
  constexpr int CINP = CIN + 8;   // padded LDS channel stride (16B multiple)
  constexpr int TW = 34;          // 32-pixel segment + 1-pixel halo each side
  constexpr int NCB = CIN / 32;   // K chunks per tap
  __shared__ bf16 xs[4 * TW * CINP];

  const int wtiles = WW / 32;
  const int w0 = (blockIdx.x % wtiles) * 32;
  const int h0 = ((blockIdx.x / wtiles) % (HH / 2)) * 2;
  const int b  = blockIdx.x / (wtiles * (HH / 2));

  // ---- cooperative halo stage: 4 rows x 34 cols x CIN channels -> bf16 LDS
  //      decode with w fastest => coalesced global reads; LDS side transposes
  //      to ci-contiguous (2-way bank conflict at worst).
  const float* inf = (const float*)in_v;
  const bf16*  inb = (const bf16*)in_v;
  const int nelem = 4 * TW * CIN;
  for (int i = threadIdx.x; i < nelem; i += 256) {
    int wc = i % TW;
    int dh = (i / TW) & 3;
    int ci = i / (TW * 4);
    int hin = h0 - 1 + dh;
    int win = w0 - 1 + wc;
    float val = 0.0f;
    if (hin >= 0 && hin < HH && win >= 0 && win < WW) {
      size_t gi = (((size_t)b * CIN + ci) * HH + hin) * WW + win;
      if constexpr (IN_F32) val = inf[gi];
      else                  val = (float)inb[gi];
    }
    xs[(dh * TW + wc) * CINP + ci] = (bf16)val;
  }
  __syncthreads();

  const int lane = threadIdx.x & 31;
  const int wave = threadIdx.x >> 5;
  const int g = lane >> 4;   // K-half group
  const int m = lane & 15;   // A-row (pixel) / B-col (cout) index
  const int cout = wave * 16 + m;

  union Frag { v16bf v; v8bf h[2]; };

  v8f acc[4] = {v8f{}, v8f{}, v8f{}, v8f{}};  // tiles: (row r, seg s) = t>>1,t&1

  // B fragments stream contiguously: 512 halves per (tap,cb) chunk per wave.
  const bf16* wp0 = wmix + (((size_t)b * 9 * NCB) * 8 + wave) * 32 * 16 +
                    (size_t)lane * 16;

#pragma unroll
  for (int tap = 0; tap < 9; ++tap) {
    const int kh = tap / 3, kw = tap % 3;
#pragma unroll
    for (int cb = 0; cb < NCB; ++cb) {
      const bf16* wp = wp0 + (size_t)(tap * NCB + cb) * (8 * 32 * 16);
      __builtin_prefetch((const void*)(wp + 8 * 32 * 16), 0, 1);
      Frag bw;
      bw.v = *(const v16bf*)wp;              // 32B contiguous per lane
      const int c0 = cb * 32 + g * 8;        // halves 0..7
      const int c1 = cb * 32 + 16 + g * 8;   // halves 8..15
#pragma unroll
      for (int t = 0; t < 4; ++t) {
        const int rr = t >> 1, ss = t & 1;
        const bf16* xr = &xs[((rr + kh) * TW + (m + ss * 16 + kw)) * CINP];
        Frag a;
        a.h[0] = *(const v8bf*)(xr + c0);    // ds_load_b128
        a.h[1] = *(const v8bf*)(xr + c1);
        acc[t] = __builtin_amdgcn_wmma_f32_16x16x32_bf16(
            false, a.v, false, bw.v, (short)0, acc[t], false, false);
      }
    }
  }

  // ---- fused bias + BatchNorm + ReLU epilogue
  const float sc = bng[cout] * rsqrtf(bnv[cout] + BN_EPS);
  const float sh = bnb[cout] - bnm[cout] * sc + bias[cout] * sc;

#pragma unroll
  for (int t = 0; t < 4; ++t) {
    const int rr = t >> 1, ss = t & 1;
    // lane writes 8 consecutive pixels starting at w0 + ss*16 + g*8
    const size_t obase = (((size_t)b * COUT + cout) * HH + h0 + rr) * WW +
                         w0 + ss * 16 + g * 8;
    if constexpr (OUT_BF16) {
      bf16* out = (bf16*)out_v;
      v8bf r;
#pragma unroll
      for (int v = 0; v < 8; ++v)
        r[v] = (bf16)fmaxf(acc[t][v] * sc + sh, 0.0f);
      *(v8bf*)(out + obase) = r;             // 16B store
    } else {
      float* out = (float*)out_v;
      v8f r;
#pragma unroll
      for (int v = 0; v < 8; ++v)
        r[v] = fmaxf(acc[t][v] * sc + sh, 0.0f);
      *(v4f*)(out + obase)     = ((v4f*)&r)[0];
      *(v4f*)(out + obase + 4) = ((v4f*)&r)[1];
    }
  }
}

// ---------------------------------------------------------------------------
// Workspace layout (bytes):
//   pooled1 @0 (4KB) | rw1 @4K | pooled2 @8K (8KB) | rw2 @16K
//   wmix1 @32K (2.25MB) | wmix2 (4.5MB) | hbuf bf16 (64MB)   => ~71MB total
// ---------------------------------------------------------------------------
extern "C" void kernel_launch(void* const* d_in, const int* in_sizes, int n_in,
                              void* d_out, int out_size, void* d_ws,
                              size_t ws_size, hipStream_t stream) {
  const float* x     = (const float*)d_in[0];
  const float* w1    = (const float*)d_in[1];
  const float* b1    = (const float*)d_in[2];
  const float* fc1_w = (const float*)d_in[3];
  const float* fc1_b = (const float*)d_in[4];
  const float* bn1_g = (const float*)d_in[5];
  const float* bn1_b = (const float*)d_in[6];
  const float* bn1_m = (const float*)d_in[7];
  const float* bn1_v = (const float*)d_in[8];
  const float* w2    = (const float*)d_in[9];
  const float* b2    = (const float*)d_in[10];
  const float* fc2_w = (const float*)d_in[11];
  const float* fc2_b = (const float*)d_in[12];
  const float* bn2_g = (const float*)d_in[13];
  const float* bn2_b = (const float*)d_in[14];
  const float* bn2_m = (const float*)d_in[15];
  const float* bn2_v = (const float*)d_in[16];

  char* ws = (char*)d_ws;
  float* pooled1 = (float*)(ws + 0);
  float* rw1     = (float*)(ws + 4096);
  float* pooled2 = (float*)(ws + 8192);
  float* rw2     = (float*)(ws + 16384);
  bf16*  wmix1   = (bf16*)(ws + 32768);
  bf16*  wmix2   = (bf16*)(ws + 32768 + 2359296);
  bf16*  hbuf    = (bf16*)(ws + 32768 + 2359296 + 4718592);

  const int conv_grid = BATCH * (HH / 2) * (WW / 32);  // 4096 blocks x 8 waves

  // ---- layer 1
  pool_mean_kernel<float><<<BATCH * 64, 256, 0, stream>>>(x, pooled1);
  routing_kernel<<<1, 64, 0, stream>>>(pooled1, fc1_w, fc1_b, rw1, 64);
  mix_weights_kernel<<<1024, 256, 0, stream>>>(w1, rw1, wmix1, 64);
  condconv_wmma_kernel<64, true, true><<<conv_grid, 256, 0, stream>>>(
      x, wmix1, b1, bn1_g, bn1_b, bn1_m, bn1_v, hbuf);

  // ---- layer 2
  pool_mean_kernel<bf16><<<BATCH * 128, 256, 0, stream>>>(hbuf, pooled2);
  routing_kernel<<<1, 64, 0, stream>>>(pooled2, fc2_w, fc2_b, rw2, 128);
  mix_weights_kernel<<<2048, 256, 0, stream>>>(w2, rw2, wmix2, 128);
  condconv_wmma_kernel<128, false, false><<<conv_grid, 256, 0, stream>>>(
      hbuf, wmix2, b2, bn2_g, bn2_b, bn2_m, bn2_v, d_out);
}
